// promptKT_91104846283081
// MI455X (gfx1250) — matrix operand order repaired and checked by reading, not compiled, not run on hardware
//
#include <hip/hip_runtime.h>

// ---------------- problem constants ----------------
constexpr int B_   = 32;
constexpr int S_   = 512;
constexpr int D_   = 1024;
constexpr int H_   = 16;
constexpr int L_   = 4;
constexpr int DFF_ = 4096;
constexpr int DK_  = 64;

// ---------------- WMMA types ----------------
typedef __attribute__((ext_vector_type(16))) __bf16          v16bf;
typedef __attribute__((ext_vector_type(8)))  float           v8f;
typedef __attribute__((ext_vector_type(8)))  unsigned short  su8;   // 8 bf16 bit patterns (16B)
typedef __attribute__((ext_vector_type(4)))  float           f4;

union BF16x16 {
    v16bf          v;
    su8            h[2];
    unsigned short us[16];
};

__device__ __forceinline__ unsigned short f2bf(float f) {
    unsigned int u = __float_as_uint(f);
    unsigned int r = u + 0x7fffu + ((u >> 16) & 1u);   // round-to-nearest-even
    return (unsigned short)(r >> 16);
}

__device__ __forceinline__ v8f zero8() {
    v8f z;
#pragma unroll
    for (int i = 0; i < 8; ++i) z[i] = 0.f;
    return z;
}

__device__ __forceinline__ v8f wmma_bf16(const BF16x16& a, const BF16x16& b, v8f c) {
    // D(16x16,f32) = A(16x32,bf16) * B(32x16,bf16) + C
    return __builtin_amdgcn_wmma_f32_16x16x32_bf16(false, a.v, false, b.v,
                                                   (short)0, c, false, false);
}

// ---------------- Tensor Data Mover (async 2D tile DMA -> LDS) ----------------
#if defined(__has_builtin)
#if __has_builtin(__builtin_amdgcn_tensor_load_to_lds)
#define HAVE_TDM 1
#endif
#endif
#ifndef HAVE_TDM
#define HAVE_TDM 0
#endif

#if HAVE_TDM
typedef __attribute__((ext_vector_type(4))) unsigned int u32x4;
typedef __attribute__((ext_vector_type(4))) int          i32x4;
typedef __attribute__((ext_vector_type(8))) int          i32x8;

// 2D bf16 tile DMA: global row-major (row length = stride_elems elements)
// -> LDS rows with padding so the LDS row stride becomes
// cols*2 + (pad_amount_code+1)*4 bytes. pad_interval_code: dwords-per-row
// as 2^(code+1); must equal cols/2 dwords. Tracked with TENSORcnt.
__device__ __forceinline__ void tdm_load_tile_bf16(
    const unsigned short* gsrc, unsigned lds_byte_addr,
    int cols, int rows, long long stride_elems,
    int pad_interval_code, int pad_amount_code)
{
    unsigned long long ga = (unsigned long long)gsrc;
    u32x4 g0;
    g0[0] = 1u;                                   // count=1, user descriptor
    g0[1] = lds_byte_addr;                        // D#.lds_addr
    g0[2] = (unsigned)(ga & 0xffffffffu);         // global_addr[31:0]
    g0[3] = (unsigned)((ga >> 32) & 0x1ffffffu)   // global_addr[56:32]
          | (2u << 30);                           // type = 2 ("image")
    unsigned td0 = (unsigned)stride_elems;        // tensor_dim0 (full row length)
    unsigned td1 = (unsigned)rows;                // tensor_dim1
    unsigned long long s0 = (unsigned long long)stride_elems;
    i32x8 g1;
    g1[0] = (int)((1u << 16)                      // data_size = 1 (2 bytes)
          | (1u << 20)                            // pad_enable
          | ((unsigned)pad_interval_code << 22)
          | ((unsigned)pad_amount_code << 25));
    g1[1] = (int)((td0 & 0xffffu) << 16);                       // dim0[15:0]
    g1[2] = (int)((td0 >> 16) | ((td1 & 0xffffu) << 16));       // dim0[31:16],dim1[15:0]
    g1[3] = (int)((td1 >> 16) | ((unsigned)cols << 16));        // dim1[31:16],tile_dim0
    g1[4] = (int)rows;                                          // tile_dim1 (tile_dim2=0)
    g1[5] = (int)(s0 & 0xffffffffu);                            // dim0_stride[31:0]
    g1[6] = (int)((s0 >> 32) & 0xffffu);                        // dim0_stride[47:32]
    g1[7] = 0;
    i32x4 z4 = {0, 0, 0, 0};
#if __clang_major__ >= 23
    i32x8 z8 = {0, 0, 0, 0, 0, 0, 0, 0};
    __builtin_amdgcn_tensor_load_to_lds(g0, g1, z4, z4, z8, 0);
#else
    __builtin_amdgcn_tensor_load_to_lds(g0, g1, z4, z4, 0);
#endif
}
#endif

// ---------------- kernel 1: x = q + pe ; y = qa + pe ----------------
__global__ __launch_bounds__(256)
void add_pe_kernel(const float* __restrict__ q, const float* __restrict__ qa,
                   float* __restrict__ x_f32,
                   unsigned short* __restrict__ x_bf,
                   unsigned short* __restrict__ y_bf) {
    size_t i = (size_t)blockIdx.x * 256 + threadIdx.x;
    size_t total = (size_t)B_ * S_ * D_;
    if (i >= total) return;
    int d  = (int)(i % D_);
    int s  = (int)((i / D_) % S_);
    int i2 = d >> 1;
    float div = __expf((float)(2 * i2) * (-9.210340371976184f / (float)D_)); // -ln(10000)/D
    float ang = (float)s * div;
    float pe  = (d & 1) ? __cosf(ang) : __sinf(ang);
    float xv = q[i] + pe;
    float yv = qa[i] + pe;
    x_f32[i] = xv;
    x_bf[i]  = f2bf(xv);
    y_bf[i]  = f2bf(yv);
}

// ---------------- kernel 2: tiled bf16 WMMA GEMM ----------------
// C[M,N] = A[M,K](bf16) * Bw[K,N](f32 weights, converted to bf16 in LDS) + bias
// 256 threads = 8 waves; tile BM=128, BN=64, BK=32; wave grid 4(M) x 2(N),
// each wave -> 32x32 output via 2x2 WMMA 16x16 tiles.
// A tile is staged by the Tensor Data Mover (wave 0 issues, TENSORcnt wait,
// barrier publishes); B tile staged cooperatively with f32->bf16 convert.
template <bool RELU, bool OUT_BF16>
__global__ __launch_bounds__(256)
void gemm_bf16_kernel(const unsigned short* __restrict__ A,
                      const float* __restrict__ Bw,
                      const float* __restrict__ bias,
                      void* __restrict__ Cout,
                      int M, int N, int K) {
    constexpr int BM = 128, BN = 64, BK = 32;
    constexpr int AS = BK + 8;  // 40 ushorts = 80B (16B multiple): 16dw + 4dw pad
    constexpr int BS = BK + 8;  // Bt row stride
    __shared__ unsigned short Alds[BM * AS];
    __shared__ unsigned short Blds[BN * BS];   // Bt[n][k]

    const int tid  = threadIdx.x;
    const int wid  = tid >> 5;
    const int lane = tid & 31;
    const int h    = lane >> 4;
    const int ln   = lane & 15;
    const int wm   = wid >> 1;    // 0..3
    const int wn   = wid & 1;     // 0..1
    const int m0   = blockIdx.y * BM;
    const int n0   = blockIdx.x * BN;

    v8f acc[2][2];
#pragma unroll
    for (int i = 0; i < 2; ++i)
#pragma unroll
        for (int j = 0; j < 2; ++j) acc[i][j] = zero8();

    for (int k0 = 0; k0 < K; k0 += BK) {
        // ---- stage A tile (128x32 bf16) ----
#if HAVE_TDM
        if (wid == 0) {
            // pad_interval: 16 dwords (code 3) = one 64B row; pad 4 dwords (code 3)
            tdm_load_tile_bf16(A + (size_t)m0 * K + k0,
                               (unsigned)(unsigned long long)&Alds[0],
                               BK, BM, K, /*interval=16dw*/3, /*amount=4dw*/3);
        }
#else
        {
            int row = tid >> 1, cb = (tid & 1) * 16;
            const su8* src = (const su8*)(A + (size_t)(m0 + row) * K + k0 + cb);
            su8 a0 = src[0], a1 = src[1];
            *(su8*)(&Alds[row * AS + cb])     = a0;
            *(su8*)(&Alds[row * AS + cb + 8]) = a1;
        }
#endif
        // ---- stage B tile transposed: f32 Bw[k][n] (32x64) -> bf16 Bt[n][k] ----
        {
            int kk = tid >> 3, nb = (tid & 7) * 8;
            const float* src = Bw + (size_t)(k0 + kk) * N + n0 + nb;
            f4 b0 = *(const f4*)(src);
            f4 b1 = *(const f4*)(src + 4);
            if (k0 + BK < K)   // prefetch next weight tile into GL2
                __builtin_prefetch(src + (size_t)BK * N, 0, 3);
#pragma unroll
            for (int j = 0; j < 4; ++j) Blds[(nb + j) * BS + kk]     = f2bf(b0[j]);
#pragma unroll
            for (int j = 0; j < 4; ++j) Blds[(nb + 4 + j) * BS + kk] = f2bf(b1[j]);
        }
#if HAVE_TDM
        if (wid == 0) __builtin_amdgcn_s_wait_tensorcnt(0);
#endif
        __syncthreads();

        BF16x16 a[2], b[2];
#pragma unroll
        for (int sm = 0; sm < 2; ++sm) {
            const unsigned short* ap = &Alds[(wm * 32 + sm * 16 + ln) * AS];
            a[sm].h[0] = *(const su8*)(ap + 16 * h);       // K = 16h+e
            a[sm].h[1] = *(const su8*)(ap + 16 * h + 8);   // K = 16h+e
        }
#pragma unroll
        for (int sn = 0; sn < 2; ++sn) {
            const unsigned short* bp = &Blds[(wn * 32 + sn * 16 + ln) * BS + 16 * h];
            b[sn].h[0] = *(const su8*)(bp);                // K = 16h+e
            b[sn].h[1] = *(const su8*)(bp + 8);
        }
#pragma unroll
        for (int sm = 0; sm < 2; ++sm)
#pragma unroll
            for (int sn = 0; sn < 2; ++sn)
                acc[sm][sn] = wmma_bf16(a[sm], b[sn], acc[sm][sn]);
        __syncthreads();
    }

    // epilogue: C layout lane -> (N = ln, M = r + 8h)
#pragma unroll
    for (int sm = 0; sm < 2; ++sm) {
#pragma unroll
        for (int sn = 0; sn < 2; ++sn) {
            int col  = n0 + wn * 32 + sn * 16 + ln;
            int rowb = m0 + wm * 32 + sm * 16 + 8 * h;
            float bv = bias[col];
#pragma unroll
            for (int r = 0; r < 8; ++r) {
                float v = acc[sm][sn][r] + bv;
                if (RELU) v = fmaxf(v, 0.f);
                size_t idx = (size_t)(rowb + r) * N + col;
                if (OUT_BF16) ((unsigned short*)Cout)[idx] = f2bf(v);
                else          ((float*)Cout)[idx]          = v;
            }
        }
    }
}

// ---------------- kernel 3: transposed flash attention ----------------
// grid (S/64, H, B), 128 threads = 4 waves; wave w owns 16 queries.
// Computes S^T = K*Q^T so query axis = lane axis of the WMMA C layout,
// then O^T += V^T * P^T. Strictly-causal mask (key >= query masked),
// query row 0 forced to zero (reference's zero_pad trick).
// K tile staged by TDM; V tile transposed cooperatively into LDS.
__global__ __launch_bounds__(128)
void attn_kernel(const unsigned short* __restrict__ QK,  // (B,S,D) bf16, Q==K
                 const unsigned short* __restrict__ V,   // (B,S,D) bf16
                 unsigned short* __restrict__ Obf) {     // (B,S,D) bf16
    constexpr int KS = 72;  // K-tile row stride (144B): 32dw row + 4dw pad
    constexpr int VS = 40;  // V^T row stride (80B)
    __shared__ unsigned short Kl[32 * KS];     // [key][dk]
    __shared__ unsigned short Vt[64 * VS];     // [dk][key]

    const int qb   = blockIdx.x * 64;
    const int head = blockIdx.y;
    const int b    = blockIdx.z;
    const int tid  = threadIdx.x;
    const int wid  = tid >> 5;
    const int lane = tid & 31;
    const int h    = lane >> 4;
    const int ln   = lane & 15;

    const size_t base = ((size_t)b * S_) * D_ + (size_t)head * DK_;
    const int q = qb + wid * 16 + ln;          // this lane's query (column N)

    // B-operand Q^T: element e of chunk c  <->  dk = 32c + 16h + e
    BF16x16 bq[2];
    {
        const unsigned short* qp = QK + base + (size_t)q * D_;
#pragma unroll
        for (int c = 0; c < 2; ++c) {
            bq[c].h[0] = *(const su8*)(qp + 32 * c + 16 * h);
            bq[c].h[1] = *(const su8*)(qp + 32 * c + 16 * h + 8);
        }
    }

    v8f o[4];
#pragma unroll
    for (int d = 0; d < 4; ++d) o[d] = zero8();
    float m = -1e30f, l = 0.f;

    const int ntiles = (qb + 64) / 32;         // keys [0, qb+64) cover all j < q
    for (int t = 0; t < ntiles; ++t) {
        const int kt = t * 32;
        __syncthreads();
        // ---- stage K tile rows kt..kt+31 (32x64 bf16) ----
#if HAVE_TDM
        if (wid == 0) {
            // row = 128B = 32 dwords (code 4), pad 4 dwords (code 3) -> stride 144B
            tdm_load_tile_bf16(QK + base + (size_t)kt * D_,
                               (unsigned)(unsigned long long)&Kl[0],
                               64, 32, D_, /*interval=32dw*/4, /*amount=4dw*/3);
        }
#else
        {
            int row = tid >> 2, cb = (tid & 3) * 16;
            const su8* src = (const su8*)(QK + base + (size_t)(kt + row) * D_ + cb);
            *(su8*)(&Kl[row * KS + cb])     = src[0];
            *(su8*)(&Kl[row * KS + cb + 8]) = src[1];
        }
#endif
        // ---- stage V tile transposed: Vt[dk][key] ----
        {
            int row = tid >> 2, cb = (tid & 3) * 16;
            const su8* src = (const su8*)(V + base + (size_t)(kt + row) * D_ + cb);
            su8 v0 = src[0], v1 = src[1];
#pragma unroll
            for (int j = 0; j < 8; ++j) Vt[(cb + j) * VS + row]     = v0[j];
#pragma unroll
            for (int j = 0; j < 8; ++j) Vt[(cb + 8 + j) * VS + row] = v1[j];
        }
#if HAVE_TDM
        if (wid == 0) __builtin_amdgcn_s_wait_tensorcnt(0);
#endif
        __syncthreads();

        // scores S^T: two 16(key)x16(query) tiles; A = K rows, chained over dk
        v8f st[2];
#pragma unroll
        for (int s2 = 0; s2 < 2; ++s2) {
            const unsigned short* kp = &Kl[(s2 * 16 + ln) * KS];
            v8f c = zero8();
#pragma unroll
            for (int c2 = 0; c2 < 2; ++c2) {
                BF16x16 a;
                a.h[0] = *(const su8*)(kp + 32 * c2 + 16 * h);
                a.h[1] = *(const su8*)(kp + 32 * c2 + 16 * h + 8);
                c = wmma_bf16(a, bq[c2], c);
            }
            st[s2] = c;
        }

        // scale + strictly-causal mask; per-query (per-lane) online softmax
        float lmax = -1e30f;
#pragma unroll
        for (int s2 = 0; s2 < 2; ++s2)
#pragma unroll
            for (int r = 0; r < 8; ++r) {
                int key = kt + s2 * 16 + 8 * h + r;
                float sv = st[s2][r] * 0.125f;          // 1/sqrt(64)
                if (key >= q) sv = -1e30f;
                st[s2][r] = sv;
                lmax = fmaxf(lmax, sv);
            }
        lmax = fmaxf(lmax, __shfl_xor(lmax, 16));       // other half holds other keys
        float mnew = fmaxf(m, lmax);
        float corr = __expf(m - mnew);
        m = mnew;

        float p[2][8];
        float lsum = 0.f;
#pragma unroll
        for (int s2 = 0; s2 < 2; ++s2)
#pragma unroll
            for (int r = 0; r < 8; ++r) {
                float pv = __expf(st[s2][r] - m);
                p[s2][r] = pv;
                lsum += pv;
            }
        lsum += __shfl_xor(lsum, 16);
        l = l * corr + lsum;
#pragma unroll
        for (int d = 0; d < 4; ++d)
#pragma unroll
            for (int r = 0; r < 8; ++r) o[d][r] *= corr;

        // build P^T B-operand (32 keys x 16 queries): element e <-> key 16h+e
        BF16x16 pb;
#pragma unroll
        for (int e = 0; e < 8; ++e) {
            float x0 = p[0][e], x1 = p[1][e];
            float x0sw = __shfl_xor(x0, 16);
            float x1sw = __shfl_xor(x1, 16);
            pb.us[e]     = f2bf(h ? x1sw : x0);  // keys 16h + e
            pb.us[e + 8] = f2bf(h ? x1  : x0sw); // keys 16h + 8 + e
        }

        // O^T += V^T * P^T (4 dk chunks)
#pragma unroll
        for (int d = 0; d < 4; ++d) {
            const unsigned short* vp = &Vt[(d * 16 + ln) * VS];
            BF16x16 av;
            av.h[0] = *(const su8*)(vp + 16 * h);
            av.h[1] = *(const su8*)(vp + 16 * h + 8);
            o[d] = wmma_bf16(av, pb, o[d]);
        }
    }

    // normalize + store; lane owns query q, dk = 16d + 8h + r (contiguous in r)
    float invl = (q == 0) ? 0.f : 1.f / l;     // zero_pad: attn row 0 -> 0
#pragma unroll
    for (int d = 0; d < 4; ++d) {
        su8 ov;
#pragma unroll
        for (int r = 0; r < 8; ++r) ov[r] = f2bf(o[d][r] * invl);
        *(su8*)(Obf + base + (size_t)q * D_ + d * 16 + 8 * h) = ov;
    }
}

// ---------------- kernel 4: residual + LayerNorm ----------------
// one block (256 threads) per row of D=1024; writes fp32 stream + bf16 copy
__global__ __launch_bounds__(256)
void resid_ln_kernel(const float* __restrict__ base, const float* __restrict__ delta,
                     const float* __restrict__ g, const float* __restrict__ be,
                     float* __restrict__ out_f32, unsigned short* __restrict__ out_bf) {
    __shared__ float red[8];
    const size_t row = blockIdx.x;
    const float* bp = base  + row * D_;
    const float* dp = delta + row * D_;
    const int lane = threadIdx.x & 31, wid = threadIdx.x >> 5;

    float v[4], s = 0.f;
#pragma unroll
    for (int j = 0; j < 4; ++j) {
        int d = threadIdx.x + j * 256;
        v[j] = bp[d] + dp[d];
        s += v[j];
    }
#pragma unroll
    for (int off = 16; off; off >>= 1) s += __shfl_xor(s, off);
    if (lane == 0) red[wid] = s;
    __syncthreads();
    float tot = 0.f;
#pragma unroll
    for (int i = 0; i < 8; ++i) tot += red[i];
    float mean = tot * (1.f / D_);

    float vs = 0.f;
#pragma unroll
    for (int j = 0; j < 4; ++j) { float c = v[j] - mean; vs += c * c; }
#pragma unroll
    for (int off = 16; off; off >>= 1) vs += __shfl_xor(vs, off);
    __syncthreads();
    if (lane == 0) red[wid] = vs;
    __syncthreads();
    float tot2 = 0.f;
#pragma unroll
    for (int i = 0; i < 8; ++i) tot2 += red[i];
    float rstd = rsqrtf(tot2 * (1.f / D_) + 1e-5f);

#pragma unroll
    for (int j = 0; j < 4; ++j) {
        int d = threadIdx.x + j * 256;
        float o = (v[j] - mean) * rstd * g[d] + be[d];
        out_f32[row * D_ + d] = o;
        out_bf[row * D_ + d]  = f2bf(o);
    }
}

// ---------------- host launcher ----------------
extern "C" void kernel_launch(void* const* d_in, const int* in_sizes, int n_in,
                              void* d_out, int out_size, void* d_ws, size_t ws_size,
                              hipStream_t stream) {
    (void)in_sizes; (void)n_in; (void)out_size; (void)ws_size;

    const float* q_embed = (const float*)d_in[0];
    const float* qa_embed= (const float*)d_in[1];
    const float* Wk  = (const float*)d_in[2];
    const float* bk  = (const float*)d_in[3];
    const float* Wv  = (const float*)d_in[4];
    const float* bv  = (const float*)d_in[5];
    const float* Wo  = (const float*)d_in[6];
    const float* bo  = (const float*)d_in[7];
    const float* g1  = (const float*)d_in[8];
    const float* be1 = (const float*)d_in[9];
    const float* W1  = (const float*)d_in[10];
    const float* c1  = (const float*)d_in[11];
    const float* W2  = (const float*)d_in[12];
    const float* c2  = (const float*)d_in[13];
    const float* g2  = (const float*)d_in[14];
    const float* be2 = (const float*)d_in[15];
    float* out = (float*)d_out;

    // workspace carving (~416 MB total)
    char* ws = (char*)d_ws;
    auto alloc = [&](size_t bytes) {
        char* p = ws;
        ws += (bytes + 255) & ~(size_t)255;
        return p;
    };
    const size_t ND = (size_t)B_ * S_ * D_;
    float*          x_f32    = (float*)alloc(ND * 4);
    float*          gemm_f32 = (float*)alloc(ND * 4);
    unsigned short* x_bf     = (unsigned short*)alloc(ND * 2);
    unsigned short* y_bf     = (unsigned short*)alloc(ND * 2);
    unsigned short* qk_bf    = (unsigned short*)alloc(ND * 2);
    unsigned short* v_bf     = (unsigned short*)alloc(ND * 2);
    unsigned short* ao_bf    = (unsigned short*)alloc(ND * 2);
    unsigned short* ff_bf    = (unsigned short*)alloc((size_t)B_ * S_ * DFF_ * 2);

    add_pe_kernel<<<dim3((unsigned)((ND + 255) / 256)), 256, 0, stream>>>(
        q_embed, qa_embed, x_f32, x_bf, y_bf);

    const int Mrows = B_ * S_;
    const dim3 gD(D_ / 64, Mrows / 128);      // N=1024 GEMMs
    const dim3 gF(DFF_ / 64, Mrows / 128);    // N=4096 GEMM
    const dim3 gA(S_ / 64, H_, B_);           // attention

    for (int l = 0; l < L_; ++l) {
        const float* Wk_l = Wk + (size_t)l * D_ * D_;
        const float* bk_l = bk + (size_t)l * D_;
        const float* Wv_l = Wv + (size_t)l * D_ * D_;
        const float* bv_l = bv + (size_t)l * D_;
        const float* Wo_l = Wo + (size_t)l * D_ * D_;
        const float* bo_l = bo + (size_t)l * D_;
        const float* g1_l = g1 + (size_t)l * D_;
        const float* be1_l= be1+ (size_t)l * D_;
        const float* W1_l = W1 + (size_t)l * D_ * DFF_;
        const float* c1_l = c1 + (size_t)l * DFF_;
        const float* W2_l = W2 + (size_t)l * DFF_ * D_;
        const float* c2_l = c2 + (size_t)l * D_;
        const float* g2_l = g2 + (size_t)l * D_;
        const float* be2_l= be2+ (size_t)l * D_;

        // Q(==K) and V projections -> bf16
        gemm_bf16_kernel<false, true><<<gD, 256, 0, stream>>>(
            x_bf, Wk_l, bk_l, qk_bf, Mrows, D_, D_);
        gemm_bf16_kernel<false, true><<<gD, 256, 0, stream>>>(
            y_bf, Wv_l, bv_l, v_bf, Mrows, D_, D_);

        attn_kernel<<<gA, 128, 0, stream>>>(qk_bf, v_bf, ao_bf);

        // output projection -> f32, then residual + LN1
        gemm_bf16_kernel<false, false><<<gD, 256, 0, stream>>>(
            ao_bf, Wo_l, bo_l, gemm_f32, Mrows, D_, D_);
        resid_ln_kernel<<<Mrows, 256, 0, stream>>>(
            x_f32, gemm_f32, g1_l, be1_l, x_f32, x_bf);

        // FFN: relu(x@W1+c1) @ W2 + c2, residual + LN2
        gemm_bf16_kernel<true, true><<<gF, 256, 0, stream>>>(
            x_bf, W1_l, c1_l, ff_bf, Mrows, DFF_, D_);
        gemm_bf16_kernel<false, false><<<gD, 256, 0, stream>>>(
            ff_bf, W2_l, c2_l, gemm_f32, Mrows, D_, DFF_);

        float* dst = (l == L_ - 1) ? out : x_f32;
        resid_ln_kernel<<<Mrows, 256, 0, stream>>>(
            x_f32, gemm_f32, g2_l, be2_l, dst, x_bf);
    }
}
